// LSTMnetwork_13580686590591
// MI455X (gfx1250) — matrix operand order repaired and checked by reading, not compiled, not run on hardware
//
#include <hip/hip_runtime.h>

// LSTM-as-GEMM for MI455X (gfx1250).
// B=64, C=128, IN=1024, H=512  ->  M=8192 rows, K=1024, 3 gates (i,g,o; f is dead
// since c0=0) * 2 directions * 512 units = 3072 packed output columns per layer.
// Workspace needed: ~50.1 MB (Wpack 18.9MB + bias + Xbf 16MB + Hbuf 16MB).

#define MM   8192      // B*C rows
#define KK   1024      // input features per layer (IN == 2H == 1024)
#define NPK  3072      // packed gate columns per layer (2 dir * 3 gates * 512)

typedef float          __attribute__((ext_vector_type(4)))  f32x4;
typedef float          __attribute__((ext_vector_type(8)))  v8f;
typedef __bf16         __attribute__((ext_vector_type(16))) v16bf;
typedef unsigned short __attribute__((ext_vector_type(4)))  u16x4;

struct F4x2 { f32x4 lo, hi; };

__device__ __forceinline__ unsigned short f2bf(float f) {
  unsigned int u = __float_as_uint(f);
  u += 0x7FFFu + ((u >> 16) & 1u);          // round-to-nearest-even
  return (unsigned short)(u >> 16);
}
__device__ __forceinline__ float sigmoidf_(float x) {
  return 1.0f / (1.0f + __expf(-x));
}

// ---------------------------------------------------------------- cast x -> bf16
__global__ __launch_bounds__(256)
void cvt_x_kernel(const float* __restrict__ x, unsigned short* __restrict__ o) {
  size_t i = ((size_t)blockIdx.x * 256 + threadIdx.x) * 4;
  f32x4 v = *(const f32x4*)(x + i);
  u16x4 r;
  r.x = f2bf(v.x); r.y = f2bf(v.y); r.z = f2bf(v.z); r.w = f2bf(v.w);
  *(u16x4*)(o + i) = r;
}

// ------------------------------------------- pack weights (gate-interleaved) + bias
// Packed column P in [0, 3072) per layer:
//   d    = P / 1536
//   p    = P % 1536 = ublk*48 + g3*16 + ulo   (g3: 0=i, 1=g, 2=o)
//   unit = ublk*16 + ulo,  source row r0 = {0,2,3}[g3]*512 + unit
__global__ __launch_bounds__(256)
void prep_w_kernel(const float* __restrict__ W_ih0,  const float* __restrict__ b_ih0,
                   const float* __restrict__ b_hh0,  const float* __restrict__ W_ih12,
                   const float* __restrict__ b_ih12, const float* __restrict__ b_hh12,
                   unsigned short* __restrict__ Wp,  float* __restrict__ biasp) {
  const int pr = blockIdx.x;            // [0, 3*3072)
  const int l  = pr / NPK;
  const int P  = pr - l * NPK;
  const int d  = P / 1536;
  const int p  = P - d * 1536;
  const int ublk = p / 48;
  const int rem  = p - ublk * 48;
  const int g3   = rem >> 4;
  const int ulo  = rem & 15;
  const int u    = ublk * 16 + ulo;
  const int gsrc = (g3 == 0) ? 0 : ((g3 == 1) ? 2 : 3);   // i, g, o of PyTorch i,f,g,o
  const int r0   = gsrc * 512 + u;

  const float* wsrc; float bsum;
  if (l == 0) {
    size_t off = (size_t)d * 2048 + r0;
    wsrc = W_ih0 + off * KK;
    bsum = b_ih0[off] + b_hh0[off];
  } else {
    size_t off = ((size_t)(l - 1) * 2 + d) * 2048 + r0;
    wsrc = W_ih12 + off * KK;
    bsum = b_ih12[off] + b_hh12[off];
  }
  int c = threadIdx.x * 4;
  f32x4 v = *(const f32x4*)(wsrc + c);
  u16x4 r;
  r.x = f2bf(v.x); r.y = f2bf(v.y); r.z = f2bf(v.z); r.w = f2bf(v.w);
  *(u16x4*)(Wp + (size_t)pr * KK + c) = r;
  if (threadIdx.x == 0) biasp[pr] = bsum;
}

// ------------------------------------------------------- GEMM + LSTM activation
// C[M, NPK] = A[M,K](bf16) * Wpack[NPK,K]^T (bf16), f32 accum via
// v_wmma_f32_16x16x32_bf16, then h = sigmoid(o)*tanh(sigmoid(i)*tanh(g)).
// Block tile 128x96, 8 waves as 4(M) x 2(N), wave tile 32x48 = 2x3 WMMA tiles.
__global__ __launch_bounds__(256)
void lstm_gemm_kernel(const unsigned short* __restrict__ A,
                      const unsigned short* __restrict__ W,
                      const float* __restrict__ bias,
                      unsigned short* __restrict__ Hout,   // bf16 out (non-final)
                      float* __restrict__ Fout,            // fp32 out (final layer)
                      int final_layer) {
  constexpr int BM = 128, BN = 96, NKSTEP = KK / 32;
  constexpr int ROWB = 80;                    // 64B of bf16 + 16B pad (bank-friendly)
  constexpr int BOFF = BM * ROWB;             // B-tile offset inside one buffer
  constexpr int BUFB = (BM + BN) * ROWB;      // 17920 B per buffer

  __shared__ __align__(16) unsigned char lds[2 * BUFB];

  const int t      = threadIdx.x;
  const int mblk   = blockIdx.x;
  const int nblk   = blockIdx.y;
  const int lane   = t & 31;
  const int wave   = t >> 5;
  const int wm     = wave & 3;                // 4 waves along M, 32 rows each
  const int wn     = wave >> 2;               // 2 waves along N, 48 cols each
  const int laneLo = lane & 15;
  const int laneHi = lane >> 4;
  const size_t m0  = (size_t)mblk * BM;
  const int    n0  = nblk * BN;

  const int trow = t >> 2;                    // 0..63: tile row handled this pass
  const int tc16 = (t & 3) * 16;              // 16B chunk within 64B tile row

  const char* gA = (const char*)A + (m0 + (size_t)trow) * (KK * 2) + tc16;
  const char* gB = (const char*)W + ((size_t)n0 + trow) * (KK * 2) + tc16;

  const v8f vzero = {0.f, 0.f, 0.f, 0.f, 0.f, 0.f, 0.f, 0.f};
  v8f acc[2][3];
#pragma unroll
  for (int i = 0; i < 2; ++i)
#pragma unroll
    for (int j = 0; j < 3; ++j) acc[i][j] = vzero;

  f32x4 ra0, ra1, rb0;
  f32x4 rb1 = {0.f, 0.f, 0.f, 0.f};

  auto g2r = [&](int kt) {                    // global -> regs (next K tile)
    const int kb = kt * 64;
    ra0 = *(const f32x4*)(gA + kb);
    ra1 = *(const f32x4*)(gA + kb + (size_t)64 * (KK * 2));
    rb0 = *(const f32x4*)(gB + kb);
    if (t < 128) rb1 = *(const f32x4*)(gB + kb + (size_t)64 * (KK * 2));
  };
  auto r2l = [&](int buf) {                   // regs -> LDS
    char* base = (char*)lds + buf * BUFB;
    *(f32x4*)(base + trow * ROWB + tc16)              = ra0;
    *(f32x4*)(base + (trow + 64) * ROWB + tc16)       = ra1;
    *(f32x4*)(base + BOFF + trow * ROWB + tc16)       = rb0;
    if (t < 128)
      *(f32x4*)(base + BOFF + (trow + 64) * ROWB + tc16) = rb1;
  };
  auto compute = [&](int buf) {               // LDS frags -> 6 WMMAs
    const char* base = (const char*)lds + buf * BUFB;
    v16bf af[2], bfr[3];
#pragma unroll
    for (int tm = 0; tm < 2; ++tm) {
      // A 16x32 bf16 frag: lane holds M=laneLo, K = 8*laneHi+[0..7] and +16
      const char* pa = base + (wm * 32 + tm * 16 + laneLo) * ROWB + laneHi * 16;
      F4x2 tmp = { *(const f32x4*)pa, *(const f32x4*)(pa + 32) };
      af[tm] = __builtin_bit_cast(v16bf, tmp);
    }
#pragma unroll
    for (int tn = 0; tn < 3; ++tn) {
      // B 32x16 bf16 frag: lane holds N=laneLo, K = 8*laneHi+[0..7] and +16
      const char* pb = base + BOFF + (wn * 48 + tn * 16 + laneLo) * ROWB + laneHi * 16;
      F4x2 tmp = { *(const f32x4*)pb, *(const f32x4*)(pb + 32) };
      bfr[tn] = __builtin_bit_cast(v16bf, tmp);
    }
#pragma unroll
    for (int tm = 0; tm < 2; ++tm)
#pragma unroll
      for (int tn = 0; tn < 3; ++tn)
        acc[tm][tn] = __builtin_amdgcn_wmma_f32_16x16x32_bf16(
            false, af[tm], false, bfr[tn], (short)0, acc[tm][tn], false, false);
  };

  g2r(0);
  r2l(0);
  __syncthreads();
  for (int kt = 0; kt < NKSTEP; ++kt) {
    if (kt + 1 < NKSTEP) g2r(kt + 1);
    compute(kt & 1);
    if (kt + 1 < NKSTEP) r2l((kt + 1) & 1);
    __syncthreads();
  }

  // ---- epilogue: gate math, fully lane-local thanks to gate-interleaved packing
  float bi[3];
#pragma unroll
  for (int tn = 0; tn < 3; ++tn)
    bi[tn] = bias[n0 + wn * 48 + tn * 16 + laneLo];

  const int d    = nblk >> 4;                  // 16 n-blocks (1536 cols) per direction
  const int ublk = (nblk & 15) * 2 + wn;       // 16-unit block index
  const int ocol = d * 512 + ublk * 16 + laneLo;
  const size_t row0 = m0 + wm * 32 + laneHi * 8;

#pragma unroll
  for (int tm = 0; tm < 2; ++tm) {
#pragma unroll
    for (int r = 0; r < 8; ++r) {
      float gi = acc[tm][0][r] + bi[0];
      float gg = acc[tm][1][r] + bi[1];
      float go = acc[tm][2][r] + bi[2];
      float cc = sigmoidf_(gi) * tanhf(gg);
      float h  = sigmoidf_(go) * tanhf(cc);
      size_t row = row0 + tm * 16 + r;         // C/D layout: M = r + 8*laneHi
      if (final_layer) Fout[row * 1024 + ocol] = h;
      else             Hout[row * 1024 + ocol] = f2bf(h);
    }
  }
}

// ---------------------------------------------------------------- launch
extern "C" void kernel_launch(void* const* d_in, const int* in_sizes, int n_in,
                              void* d_out, int out_size, void* d_ws, size_t ws_size,
                              hipStream_t stream) {
  const float* x      = (const float*)d_in[0];
  const float* W_ih0  = (const float*)d_in[1];
  const float* b_ih0  = (const float*)d_in[3];
  const float* b_hh0  = (const float*)d_in[4];
  const float* W_ih12 = (const float*)d_in[5];
  const float* b_ih12 = (const float*)d_in[7];
  const float* b_hh12 = (const float*)d_in[8];

  char* ws = (char*)d_ws;
  const size_t WPB = (size_t)3 * NPK * KK * 2;           // 18,874,368 B packed weights
  unsigned short* Wp    = (unsigned short*)ws;
  float*          biasp = (float*)(ws + WPB);            // 3*3072 f32 (36 KB)
  unsigned short* Xbf   = (unsigned short*)(ws + WPB + 65536);
  unsigned short* Hbuf  = Xbf + (size_t)MM * KK;         // second bf16 activation buf

  cvt_x_kernel<<<(MM * KK) / 1024, 256, 0, stream>>>(x, Xbf);
  prep_w_kernel<<<3 * NPK, 256, 0, stream>>>(W_ih0, b_ih0, b_hh0,
                                             W_ih12, b_ih12, b_hh12, Wp, biasp);

  dim3 grid(MM / 128, NPK / 96), blk(256);
  // layer 0: Xbf -> Hbuf ; layer 1: Hbuf -> Xbf (reuse) ; layer 2: Xbf -> d_out (f32)
  lstm_gemm_kernel<<<grid, blk, 0, stream>>>(Xbf,  Wp,                     biasp,
                                             Hbuf, nullptr, 0);
  lstm_gemm_kernel<<<grid, blk, 0, stream>>>(Hbuf, Wp + (size_t)NPK * KK,  biasp + NPK,
                                             Xbf,  nullptr, 0);
  lstm_gemm_kernel<<<grid, blk, 0, stream>>>(Xbf,  Wp + (size_t)2 * NPK * KK, biasp + 2 * NPK,
                                             nullptr, (float*)d_out, 1);
}
